// NMPLayer_40484361732764
// MI455X (gfx1250) — matrix operand order repaired and checked by reading, not compiled, not run on hardware
//
#include <hip/hip_runtime.h>

#define N_NODES  200000
#define IN_SIZE  256
#define FT_SIZE  512
#define N_GRAPHS 512

#define TILE_M 128
#define TILE_N 32                       // feature columns per ft-block
#define KSTEP  32
#define NUM_FB (FT_SIZE / TILE_N)       // 16
#define NUM_KS (IN_SIZE / KSTEP)        // 8

typedef __attribute__((ext_vector_type(16))) __bf16 v16bf;
typedef __attribute__((ext_vector_type(8)))  __bf16 v8bf;
typedef __attribute__((ext_vector_type(8)))  float  v8f;

union FragBf { v16bf v; v8bf h[2]; };

// f32 -> (bf16 hi, bf16 lo) split: hi is exact truncation, lo captures the
// next 8 mantissa bits.  hi*hi + hi*lo + lo*hi recovers ~2^-16 rel. accuracy.
__device__ __forceinline__ void bf16_split(float f, unsigned short& hi, unsigned short& lo) {
  unsigned u  = __float_as_uint(f);
  unsigned hu = u & 0xffff0000u;
  hi = (unsigned short)(hu >> 16);
  float rem = f - __uint_as_float(hu);
  lo = (unsigned short)(__float_as_uint(rem) >> 16);
}

__global__ void zero_out_kernel(float* __restrict__ out, int n) {
  int i = blockIdx.x * blockDim.x + threadIdx.x;
  if (i < n) out[i] = 0.0f;
}

__global__ __launch_bounds__(256)
void nmp_fused_kernel(const float* __restrict__ x,
                      const float* __restrict__ w,
                      const float* __restrict__ bias,
                      const int*   __restrict__ seg,
                      float*       __restrict__ out) {
  __shared__ unsigned short sAhi[TILE_M][IN_SIZE];   // 64 KB
  __shared__ unsigned short sAlo[TILE_M][IN_SIZE];   // 64 KB
  __shared__ unsigned short sBhi[TILE_N][IN_SIZE];   // 16 KB (full-K W slab)
  __shared__ unsigned short sBlo[TILE_N][IN_SIZE];   // 16 KB
  __shared__ float          sRed[TILE_M][TILE_N];    // 16 KB
  __shared__ int            sSeg[TILE_M];            // 0.5 KB => ~176.5 KB total

  const int tid      = threadIdx.x;
  const int nodeBase = blockIdx.x * TILE_M;

  // ---- Stage x tile once: f32 -> bf16 hi/lo planes in LDS ----
  for (int e = tid * 4; e < TILE_M * IN_SIZE; e += 256 * 4) {
    int row = e >> 8;            // IN_SIZE == 256
    int col = e & 255;
    int gm  = nodeBase + row;
    float4 v = make_float4(0.f, 0.f, 0.f, 0.f);
    if (gm < N_NODES) v = *(const float4*)(x + (size_t)gm * IN_SIZE + col);
    unsigned short h0,h1,h2,h3,l0,l1,l2,l3;
    bf16_split(v.x,h0,l0); bf16_split(v.y,h1,l1);
    bf16_split(v.z,h2,l2); bf16_split(v.w,h3,l3);
    *(uint2*)&sAhi[row][col] = make_uint2((unsigned)h0 | ((unsigned)h1<<16),
                                          (unsigned)h2 | ((unsigned)h3<<16));
    *(uint2*)&sAlo[row][col] = make_uint2((unsigned)l0 | ((unsigned)l1<<16),
                                          (unsigned)l2 | ((unsigned)l3<<16));
  }
  if (tid < TILE_M) {
    int gm = nodeBase + tid;
    sSeg[tid] = (gm < N_NODES) ? seg[gm] : -1;
  }

  const int lane  = tid & 31;
  const int wave  = tid >> 5;
  const int waveM = wave >> 1;       // 0..3 -> 32 rows each
  const int waveN = wave & 1;        // 0..1 -> 16 cols each
  const int lan16 = lane & 15;
  const int lhalf = lane >> 4;

  for (int fb = 0; fb < NUM_FB; ++fb) {
    // ---- Stage full-K W slab for this feature block (32 x 256) ----
    // (W is [feature][k] row-major already; each element read once per WG,
    //  and the whole 0.5 MB W matrix lives in the 192 MB L2 across WGs.)
    for (int e = tid * 4; e < TILE_N * IN_SIZE; e += 256 * 4) {
      int row = e >> 8;
      int col = e & 255;
      float4 wv = *(const float4*)(w + (size_t)(fb*TILE_N + row) * IN_SIZE + col);
      unsigned short h0,h1,h2,h3,l0,l1,l2,l3;
      bf16_split(wv.x,h0,l0); bf16_split(wv.y,h1,l1);
      bf16_split(wv.z,h2,l2); bf16_split(wv.w,h3,l3);
      *(uint2*)&sBhi[row][col] = make_uint2((unsigned)h0 | ((unsigned)h1<<16),
                                            (unsigned)h2 | ((unsigned)h3<<16));
      *(uint2*)&sBlo[row][col] = make_uint2((unsigned)l0 | ((unsigned)l1<<16),
                                            (unsigned)l2 | ((unsigned)l3<<16));
    }
    __syncthreads();

    v8f acc[2];
    #pragma unroll
    for (int mi = 0; mi < 2; ++mi)
      #pragma unroll
      for (int i = 0; i < 8; ++i) acc[mi][i] = 0.0f;

    // ---- Barrier-free K loop: 8 steps x 6 WMMAs, frags from LDS ----
    for (int ks = 0; ks < NUM_KS; ++ks) {
      // B fragment (32x16): lanes 0-15 carry K 0..15, lanes 16-31 carry K 16..31.
      FragBf bh, bl;
      {
        int brow = waveN * 16 + lan16;
        int kb   = ks * KSTEP + lhalf * 16;
        bh.h[0] = *(const v8bf*)&sBhi[brow][kb];
        bh.h[1] = *(const v8bf*)&sBhi[brow][kb + 8];
        bl.h[0] = *(const v8bf*)&sBlo[brow][kb];
        bl.h[1] = *(const v8bf*)&sBlo[brow][kb + 8];
      }
      #pragma unroll
      for (int mi = 0; mi < 2; ++mi) {
        // A fragment (16x32): per lane, 16 consecutive K at lhalf*16.
        int ar = waveM * 32 + mi * 16 + lan16;
        int ka = ks * KSTEP + lhalf * 16;
        FragBf ah, al;
        ah.h[0] = *(const v8bf*)&sAhi[ar][ka];
        ah.h[1] = *(const v8bf*)&sAhi[ar][ka + 8];
        al.h[0] = *(const v8bf*)&sAlo[ar][ka];
        al.h[1] = *(const v8bf*)&sAlo[ar][ka + 8];
        acc[mi] = __builtin_amdgcn_wmma_f32_16x16x32_bf16(false, ah.v, false, bh.v,
                                                          (short)0, acc[mi], false, false);
        acc[mi] = __builtin_amdgcn_wmma_f32_16x16x32_bf16(false, ah.v, false, bl.v,
                                                          (short)0, acc[mi], false, false);
        acc[mi] = __builtin_amdgcn_wmma_f32_16x16x32_bf16(false, al.v, false, bh.v,
                                                          (short)0, acc[mi], false, false);
      }
    }
    __syncthreads();

    // ---- Epilogue: +bias per node, then sorted-segment reduction ----
    {
      // C/D layout: lane L, vgpr r -> M = r + (L>>4)*8, N = L&15
      int col = waveN * 16 + lan16;
      float bv = bias[fb * TILE_N + col];
      #pragma unroll
      for (int mi = 0; mi < 2; ++mi) {
        int rbase = waveM * 32 + mi * 16 + lhalf * 8;
        #pragma unroll
        for (int i = 0; i < 8; ++i)
          sRed[rbase + i][col] = acc[mi][i] + bv;
      }
    }
    __syncthreads();
    {
      // Run-length scan down the sorted segment ids: one atomic per run.
      int col  = tid & 31;
      int q    = tid >> 5;                 // 8 chunks of 16 rows
      int gcol = fb * TILE_N + col;
      float run = 0.0f;
      int   cur = -1;
      for (int r = q * 16; r < q * 16 + 16; ++r) {
        int s = sSeg[r];
        if (s != cur) {
          if (cur >= 0) atomicAdd(out + (size_t)cur * FT_SIZE + gcol, run);
          run = 0.0f; cur = s;
        }
        if (s >= 0) run += sRed[r][col];
      }
      if (cur >= 0) atomicAdd(out + (size_t)cur * FT_SIZE + gcol, run);
    }
    __syncthreads();
  }
}

extern "C" void kernel_launch(void* const* d_in, const int* in_sizes, int n_in,
                              void* d_out, int out_size, void* d_ws, size_t ws_size,
                              hipStream_t stream) {
  (void)in_sizes; (void)n_in; (void)d_ws; (void)ws_size;
  const float* x    = (const float*)d_in[0];
  const float* wgt  = (const float*)d_in[1];
  const float* bias = (const float*)d_in[2];
  const int*   seg  = (const int*)d_in[3];
  float* out = (float*)d_out;

  zero_out_kernel<<<(out_size + 255) / 256, 256, 0, stream>>>(out, out_size);

  int nBlocks = (N_NODES + TILE_M - 1) / TILE_M;   // 1563
  nmp_fused_kernel<<<nBlocks, 256, 0, stream>>>(x, wgt, bias, seg, out);
}
